// QuantumDMRGLayer_26783416058516
// MI455X (gfx1250) — matrix operand order repaired and checked by reading, not compiled, not run on hardware
//
#include <hip/hip_runtime.h>
#include <hip/hip_bf16.h>

typedef __attribute__((ext_vector_type(16))) _Float16 v16h;
typedef __attribute__((ext_vector_type(8)))  _Float16 v8h;
typedef __attribute__((ext_vector_type(8)))  float    v8f;

#define BATCH 2048
#define LSITES 196
#define MBOND 16
#define NBL 10
#define NMATS (LSITES - 2)      // 194 interior sites
#define POSL 98                 // pos_label (fixed by reference)
#define HALVES_PER_SITE 512     // 32 lanes * 16 halves (1024 bytes/site)

#if __has_builtin(__builtin_amdgcn_wmma_f16_16x16x32_f16)
#define HAVE_F16D 1
#else
#define HAVE_F16D 0
#endif

// K-dimension permutation shared by the A prepack and the B construction:
//   d(K) = K>>4 ,  m(K) = (K&7) | (((K>>3)&1)<<3)
// With the C/D layout (lane s: M=0..7, lane s+16: M=8..15; f16 D packs
// M=2r,2r+1 per VGPR) the next step's B operand is entirely lane-local:
//   B[e] = (e<8 ? x0 : x1) * st[e&7].

// ---------------------------------------------------------------------------
// Pre-pass: pack A_mid (f32, (194,2,16,16)) into f16 WMMA A-operands in the
// hardware 16x32 16-bit A layout (lane l<16: row M=l, dwords j<4 -> K=2j,2j+1,
// j>=4 -> K=16+2(j-4); lanes 16..31 same with K+=8), using the permuted K map.
// Left sites (i<POSL): A_op[n][K] = A_mid[i][d][m][n]  (transposed)
// Right sites        : A_op[m][K] = A_mid[i][d][m][n]  (plain; K's m-slot = n)
// ---------------------------------------------------------------------------
__global__ __launch_bounds__(256) void dmrg_prepack(const float* __restrict__ A_mid,
                                                    _Float16* __restrict__ wsh) {
    const int i    = blockIdx.x;        // site 0..193
    const int t    = threadIdx.x;       // 0..255
    const int lane = t >> 3;            // 0..31
    const int j    = t & 7;             // dword 0..7 within lane
    const int lp   = lane & 15;         // operand row M
    const int hi   = lane >> 4;

    const int k0 = ((j < 4) ? (2 * j) : (16 + 2 * (j - 4))) + (hi ? 8 : 0);
    const int d  = j >> 2;                        // constant within the pair
    const int m0 = (hi << 3) | ((2 * j) & 7);
    const int m1 = m0 + 1;

    float v0, v1;
    if (i < POSL) { // transposed
        v0 = A_mid[(((size_t)i * 2 + d) * 16 + m0) * 16 + lp];
        v1 = A_mid[(((size_t)i * 2 + d) * 16 + m1) * 16 + lp];
    } else {        // plain (m-slot indexes the column n)
        v0 = A_mid[(((size_t)i * 2 + d) * 16 + lp) * 16 + m0];
        v1 = A_mid[(((size_t)i * 2 + d) * 16 + lp) * 16 + m1];
    }
    _Float16* dst = wsh + (size_t)i * HALVES_PER_SITE + lane * 16 + j * 2;
    dst[0] = (_Float16)v0;
    dst[1] = (_Float16)v1;
}

// One MPS site update: st' = site_matrix(x) applied to st, as a single WMMA.
__device__ __forceinline__ v8h chain_step(const float* __restrict__ xb, int site,
                                          v8h st, const _Float16* __restrict__ wsh,
                                          int lane) {
    const float2 xp = *(const float2*)(xb + 2 * (site + 1)); // (x0, x1)
    const v8h lo = st * (_Float16)xp.x;   // K block d=0  -> v_pk_mul_f16
    const v8h hb = st * (_Float16)xp.y;   // K block d=1
    const v16h Bop = __builtin_shufflevector(lo, hb, 0, 1, 2, 3, 4, 5, 6, 7,
                                             8, 9, 10, 11, 12, 13, 14, 15);
    const v16h Aop = *(const v16h*)(wsh + (size_t)site * HALVES_PER_SITE + lane * 16);
#if HAVE_F16D
    v8h c = {};
    return __builtin_amdgcn_wmma_f16_16x16x32_f16(false, Aop, false, Bop,
                                                  (short)0, c, false, false);
#else
    v8f cf = {};
    cf = __builtin_amdgcn_wmma_f32_16x16x32_f16(false, Aop, false, Bop,
                                                (short)0, cf, false, false);
    v8h r;
#pragma unroll
    for (int e = 0; e < 8; ++e) r[e] = (_Float16)cf[e];
    return r;
#endif
}

// ---------------------------------------------------------------------------
// Main kernel: one wave32 per 16-sample tile. Left and right sweeps are two
// independent WMMA dependency chains, interleaved in one loop so each WMMA's
// RAW-hazard window and operand loads are covered by the other chain.
// ---------------------------------------------------------------------------
__global__ __launch_bounds__(128) void dmrg_chain(const float* __restrict__ in,
                                                  const float* __restrict__ A_left,
                                                  const float* __restrict__ A_right,
                                                  const float* __restrict__ T_out,
                                                  const _Float16* __restrict__ wsh,
                                                  float* __restrict__ out) {
    __shared__ float Tl[MBOND * MBOND * NBL]; // 2560 floats = 10 KB
    for (int t = threadIdx.x; t < MBOND * MBOND * NBL; t += 128) Tl[t] = T_out[t];

    const int lane = threadIdx.x & 31;
    const int wave = threadIdx.x >> 5;
    const int tile = blockIdx.x * 4 + wave;     // 32 blocks * 4 waves = 128 tiles
    const int s    = lane & 15;
    const int hi   = lane >> 4;                 // 0: bond rows 0..7, 1: rows 8..15
    const int b    = tile * 16 + s;
    const float* xb = in + (size_t)b * (LSITES * 2);
    const int mbase = hi ? 8 : 0;

    v8h sL, sR;  // half-states (8 bond entries each), element e = S[mbase+e, s]

    { // boundaries: v = x0*A[0,:] + x1*A[1,:]
        const float l0 = xb[0], l1 = xb[1];
        const float r0 = xb[2 * (LSITES - 1)], r1 = xb[2 * (LSITES - 1) + 1];
#pragma unroll
        for (int r = 0; r < 8; ++r) {
            sL[r] = (_Float16)(l0 * A_left[mbase + r] + l1 * A_left[16 + mbase + r]);
            sR[r] = (_Float16)(r0 * A_right[mbase + r] + r1 * A_right[16 + mbase + r]);
        }
    }

    // interleaved sweeps: left sites 0..97, right sites 193..98 (96 steps)
    for (int t = 0; t < NMATS - POSL; ++t) {     // 96 shared iterations
        __builtin_prefetch(wsh + (size_t)(t + 1) * HALVES_PER_SITE + lane * 16, 0, 3);
        __builtin_prefetch(wsh + (size_t)(NMATS - 2 - t) * HALVES_PER_SITE + lane * 16, 0, 3);
        sL = chain_step(xb, t, sL, wsh, lane);
        sR = chain_step(xb, NMATS - 1 - t, sR, wsh, lane);
    }
    for (int t = NMATS - POSL; t < POSL; ++t) {  // left-only tail (2 steps)
        sL = chain_step(xb, t, sL, wsh, lane);
    }

    // assemble full f32 left/right vectors (one-time cross-lane exchange)
    float leftf[16], rightf[16];
    {
        float ol[8], orr[8], cl[8], cr[8];
#pragma unroll
        for (int r = 0; r < 8; ++r) { cl[r] = (float)sL[r]; cr[r] = (float)sR[r]; }
#pragma unroll
        for (int r = 0; r < 8; ++r) {
            ol[r]  = __shfl_xor(cl[r], 16, 32);
            orr[r] = __shfl_xor(cr[r], 16, 32);
        }
#pragma unroll
        for (int r = 0; r < 8; ++r) {
            leftf[r]      = hi ? ol[r] : cl[r];
            leftf[8 + r]  = hi ? cl[r] : ol[r];
            rightf[r]     = hi ? orr[r] : cr[r];
            rightf[8 + r] = hi ? cr[r] : orr[r];
        }
    }

    __syncthreads();  // Tl ready

    // ---- epilogue: out[b,l] = sum_{m,n} left[m] * T[m,n,l] * right[n] ----
    // lane halves split labels: lanes<16 -> l=0..4, lanes>=16 -> l=5..9
    float acc[5] = {0.f, 0.f, 0.f, 0.f, 0.f};
    for (int m = 0; m < 16; ++m) {
        const float lm = leftf[m];
#pragma unroll 4
        for (int n = 0; n < 16; ++n) {
            const float p = lm * rightf[n];
            const float* tp = Tl + (m * 16 + n) * NBL + hi * 5;
#pragma unroll
            for (int l = 0; l < 5; ++l) acc[l] += p * tp[l];
        }
    }
    float* ob = out + (size_t)b * NBL + hi * 5;
#pragma unroll
    for (int l = 0; l < 5; ++l) ob[l] = acc[l];
}

extern "C" void kernel_launch(void* const* d_in, const int* in_sizes, int n_in,
                              void* d_out, int out_size, void* d_ws, size_t ws_size,
                              hipStream_t stream) {
    const float* inputs  = (const float*)d_in[0];  // (2048,196,2)
    const float* A_left  = (const float*)d_in[1];  // (2,16)
    const float* A_mid   = (const float*)d_in[2];  // (194,2,16,16)
    const float* A_right = (const float*)d_in[3];  // (2,16)
    const float* T_out   = (const float*)d_in[4];  // (16,16,10)
    float* out = (float*)d_out;                    // (2048,10)
    _Float16* wsh = (_Float16*)d_ws;               // 194 KiB packed f16 operands

    dmrg_prepack<<<NMATS, 256, 0, stream>>>(A_mid, wsh);
    dmrg_chain<<<32, 128, 0, stream>>>(inputs, A_left, A_right, T_out, wsh, out);
}
